// HeteroGNN_22282290331736
// MI455X (gfx1250) — matrix operand order repaired and checked by reading, not compiled, not run on hardware
//
#include <hip/hip_runtime.h>

// ---------------------------------------------------------------------------
// Types for WMMA fragments (wave32, CDNA5 gfx1250)
// ---------------------------------------------------------------------------
typedef unsigned short u16b;
typedef __attribute__((ext_vector_type(8)))  u16b  us8;
typedef __attribute__((ext_vector_type(16))) u16b  us16;
typedef __attribute__((ext_vector_type(16))) __bf16 v16bf;
typedef __attribute__((ext_vector_type(8)))  float v8f;

// Types matching __builtin_amdgcn_global_load_async_to_lds_b128's signature:
// (v4i32 addrspace(1)*, v4i32 addrspace(3)*, imm offset, imm cpol)
typedef int v4i __attribute__((vector_size(16)));
typedef __attribute__((address_space(1))) v4i g_v4i;
typedef __attribute__((address_space(3))) v4i l_v4i;

#if defined(__has_builtin)
#if __has_builtin(__builtin_amdgcn_global_load_async_to_lds_b128)
#define HAVE_ASYNC_LDS 1
#endif
#endif
#ifndef HAVE_ASYNC_LDS
#define HAVE_ASYNC_LDS 0
#endif

__device__ __forceinline__ u16b f2bfbits(float f) {
  unsigned u = __builtin_bit_cast(unsigned, f);
  unsigned r = u + 0x7FFFu + ((u >> 16) & 1u);   // round-to-nearest-even
  return (u16b)(r >> 16);
}
__device__ __forceinline__ float bf2f(u16b b) {
  unsigned u = ((unsigned)b) << 16;
  return __builtin_bit_cast(float, u);
}

// 16-byte global -> LDS copy: async (ASYNCcnt-tracked) when available.
__device__ __forceinline__ void cp16_g2l(const u16b* g, u16b* l) {
#if HAVE_ASYNC_LDS
  __builtin_amdgcn_global_load_async_to_lds_b128((g_v4i*)g, (l_v4i*)l, 0, 0);
#else
  *(us8*)l = *(const us8*)g;
#endif
}

__device__ __forceinline__ void wait_async_lds() {
#if HAVE_ASYNC_LDS
#if __has_builtin(__builtin_amdgcn_s_wait_asynccnt)
  __builtin_amdgcn_s_wait_asynccnt(0);
#else
  asm volatile("s_wait_asynccnt 0" ::: "memory");
#endif
#endif
}

// ---------------------------------------------------------------------------
// f32 -> bf16 conversion
// ---------------------------------------------------------------------------
__global__ void f2bf_kernel(const float* __restrict__ in, u16b* __restrict__ out, int n) {
  int i = blockIdx.x * blockDim.x + threadIdx.x;
  if (i < n) out[i] = f2bfbits(in[i]);
}

__global__ void zero_kernel(float* __restrict__ p, int n) {
  int i = blockIdx.x * blockDim.x + threadIdx.x;
  if (i < n) p[i] = 0.0f;
}

// ---------------------------------------------------------------------------
// WMMA GEMM:  out[M x N] = A[M x K](bf16) @ W[N x K]^T(bf16) + bias[N]
// Block = 256 threads = 8 waves; block tile 128x64, each wave owns 16x64.
// Double-buffered LDS staging of the A panel (128xK) and shared B panel
// (64xK) via async global->LDS copies (ASYNCcnt) with K-stage = 64.
// Stage body: batch-load all 10 fragments (ds_load_b128) into registers,
// then issue 8 back-to-back WMMAs (XDL-friendly, one dscnt wait).
// Requires: M % 128 == 0, N % 64 == 0, K % 64 == 0 (true for all call sites).
// ---------------------------------------------------------------------------
#define APAD 72

__global__ __launch_bounds__(256)
void gemm_bf16_kernel(const u16b* __restrict__ A, int lda,
                      const u16b* __restrict__ W, int ldw,
                      const float* __restrict__ bias,
                      float* __restrict__ outF, u16b* __restrict__ outB, int ldo,
                      int K) {
  __shared__ __align__(16) u16b As[2][128 * APAD];
  __shared__ __align__(16) u16b Bs[2][64 * APAD];

  const int tid  = threadIdx.x;
  const int wave = tid >> 5;
  const int lane = tid & 31;
  const int h    = lane >> 4;   // half-wave select (K-split per ISA layout)
  const int l    = lane & 15;   // row (A) / col (B,C) within 16x16 tile

  const int rowBlk = blockIdx.x * 128;
  const int colBlk = blockIdx.y * 64;

  v8f acc[4];
#pragma unroll
  for (int j = 0; j < 4; ++j) {
    float bv = bias ? bias[colBlk + j * 16 + l] : 0.0f;
#pragma unroll
    for (int r = 0; r < 8; ++r) acc[j][r] = bv;
  }

  // Stage loader: copy A[128 x 64] and B[64 x 64] bf16 panels into LDS buf.
  auto loadStage = [&](int buf, int k0) {
#pragma unroll
    for (int i = 0; i < 4; ++i) {          // A: 1024 16B chunks / 256 threads
      int c   = tid + 256 * i;
      int row = c >> 3;
      int kc  = (c & 7) * 8;
      cp16_g2l(A + (size_t)(rowBlk + row) * lda + k0 + kc,
               &As[buf][row * APAD + kc]);
    }
#pragma unroll
    for (int i = 0; i < 2; ++i) {          // B: 512 16B chunks / 256 threads
      int c   = tid + 256 * i;
      int col = c >> 3;
      int kc  = (c & 7) * 8;
      cp16_g2l(W + (size_t)(colBlk + col) * ldw + k0 + kc,
               &Bs[buf][col * APAD + kc]);
    }
  };

  loadStage(0, 0);
  const int nStages = K >> 6;
  for (int s = 0; s < nStages; ++s) {
    const int cur = s & 1;
    wait_async_lds();        // async writes for stage s resident in LDS
    __syncthreads();         // all waves past stage s-1 compute
    if (s + 1 < nStages) loadStage(cur ^ 1, (s + 1) * 64);

    // Phase 1: batch-load all fragments for both k-steps of this stage.
    union Frag { us16 v; us8 p[2]; };
    Frag fa[2];
    Frag fb[2][4];
#pragma unroll
    for (int t2 = 0; t2 < 2; ++t2) {
      const int kk = t2 * 32;
      // A fragment: lane l+16h holds row l, K = kk+8h..+7 and kk+16+8h..+7
      const u16b* ab = &As[cur][(wave * 16 + l) * APAD + kk + 8 * h];
      fa[t2].p[0] = *(const us8*)ab;
      fa[t2].p[1] = *(const us8*)(ab + 16);
#pragma unroll
      for (int j = 0; j < 4; ++j) {
        // B fragment: lane n+16h holds col n, K = kk+8h..+7 and kk+16+8h..+7
        const u16b* bb = &Bs[cur][(j * 16 + l) * APAD + kk + 8 * h];
        fb[t2][j].p[0] = *(const us8*)bb;
        fb[t2][j].p[1] = *(const us8*)(bb + 16);
      }
    }
    // Phase 2: 8 back-to-back WMMAs.
#pragma unroll
    for (int t2 = 0; t2 < 2; ++t2) {
      v16bf af = __builtin_bit_cast(v16bf, fa[t2].v);
#pragma unroll
      for (int j = 0; j < 4; ++j) {
        v16bf bfv = __builtin_bit_cast(v16bf, fb[t2][j].v);
        acc[j] = __builtin_amdgcn_wmma_f32_16x16x32_bf16(
            false, af, false, bfv, (short)0, acc[j], false, false);
      }
    }
  }

  // C layout: lane n+16h, VGPR r -> row r+8h, col n
#pragma unroll
  for (int j = 0; j < 4; ++j) {
    int col = colBlk + j * 16 + l;
#pragma unroll
    for (int r = 0; r < 8; ++r) {
      int row = rowBlk + wave * 16 + r + 8 * h;
      float v = acc[j][r];
      if (outF) outF[(size_t)row * ldo + col] = v;
      if (outB) outB[(size_t)row * ldo + col] = f2bfbits(v);
    }
  }
}

// ---------------------------------------------------------------------------
// Edge scatter-add (segment_sum): one block per edge, threads over features.
// ---------------------------------------------------------------------------
__global__ void scatter_add_kernel(const float* __restrict__ x,
                                   const int* __restrict__ src,
                                   const int* __restrict__ dst,
                                   float* __restrict__ s, float* __restrict__ cnt,
                                   int F) {
  int e  = blockIdx.x;
  int si = src[e], di = dst[e];
  const float* xr = x + (size_t)si * F;
  float* sr = s + (size_t)di * F;
  for (int f = threadIdx.x; f < F; f += blockDim.x) atomicAdd(&sr[f], xr[f]);
  if (threadIdx.x == 0) atomicAdd(&cnt[di], 1.0f);
}

__global__ void finalize_mean_kernel(const float* __restrict__ s,
                                     const float* __restrict__ cnt,
                                     u16b* __restrict__ aggb, int total, int F) {
  int i = blockIdx.x * blockDim.x + threadIdx.x;
  if (i >= total) return;
  float c = cnt[i / F];
  c = c > 1.0f ? c : 1.0f;
  aggb[i] = f2bfbits(s[i] / c);
}

// ---------------------------------------------------------------------------
// Attention: z[n] = sum_a tanh(t[n,a]) * w2[a]; block-reduce, atomic into zsum
// ---------------------------------------------------------------------------
__global__ void attn_z_kernel(const float* __restrict__ t, const float* __restrict__ w2,
                              float* __restrict__ zsum, int N) {
  __shared__ float red[256];
  int n = blockIdx.x * blockDim.x + threadIdx.x;
  float z = 0.0f;
  if (n < N) {
    const float* tr = t + (size_t)n * 64;
#pragma unroll
    for (int a = 0; a < 64; ++a) z += tanhf(tr[a]) * w2[a];
  }
  red[threadIdx.x] = z;
  __syncthreads();
  for (int s = 128; s > 0; s >>= 1) {
    if ((int)threadIdx.x < s) red[threadIdx.x] += red[threadIdx.x + s];
    __syncthreads();
  }
  if (threadIdx.x == 0) atomicAdd(zsum, red[0]);
}

__global__ void attn_alpha_kernel(const float* __restrict__ zsum, float invN,
                                  float* __restrict__ alpha) {
  float z0 = zsum[0] * invN, z1 = zsum[1] * invN;
  float m  = fmaxf(z0, z1);
  float e0 = expf(z0 - m), e1 = expf(z1 - m);
  float inv = 1.0f / (e0 + e1);
  alpha[0] = e0 * inv;
  alpha[1] = e1 * inv;
}

__global__ void attn_combine_kernel(const u16b* __restrict__ e0, const u16b* __restrict__ e1,
                                    const float* __restrict__ alpha,
                                    float* __restrict__ out, int n) {
  int i = blockIdx.x * blockDim.x + threadIdx.x;
  if (i < n) out[i] = alpha[0] * bf2f(e0[i]) + alpha[1] * bf2f(e1[i]);
}

// ---------------------------------------------------------------------------
// BatchNorm (training-mode stats) + LeakyReLU
// ---------------------------------------------------------------------------
__global__ void bn_stats_kernel(const float* __restrict__ x, float* __restrict__ mean,
                                float* __restrict__ var, int N, int H) {
  __shared__ float s1[256], s2[256];
  int f = blockIdx.x;
  float a = 0.0f, b = 0.0f;
  for (int n = threadIdx.x; n < N; n += blockDim.x) {
    float v = x[(size_t)n * H + f];
    a += v; b += v * v;
  }
  s1[threadIdx.x] = a; s2[threadIdx.x] = b;
  __syncthreads();
  for (int s = 128; s > 0; s >>= 1) {
    if ((int)threadIdx.x < s) {
      s1[threadIdx.x] += s1[threadIdx.x + s];
      s2[threadIdx.x] += s2[threadIdx.x + s];
    }
    __syncthreads();
  }
  if (threadIdx.x == 0) {
    float m = s1[0] / N;
    mean[f] = m;
    var[f]  = s2[0] / N - m * m;
  }
}

__global__ void bn_apply_kernel(const float* __restrict__ x, const float* __restrict__ mean,
                                const float* __restrict__ var, const float* __restrict__ g,
                                const float* __restrict__ b, float* __restrict__ outF,
                                u16b* __restrict__ outB, int total, int H) {
  int i = blockIdx.x * blockDim.x + threadIdx.x;
  if (i >= total) return;
  int f = i & (H - 1);   // H == 256
  float y = (x[i] - mean[f]) * rsqrtf(var[f] + 1.0f) * g[f] + b[f];
  y = y >= 0.0f ? y : 0.01f * y;
  outF[i] = y;
  outB[i] = f2bfbits(y);
}

// ---------------------------------------------------------------------------
// Edge score: one wave32 per edge, H=256 dot, shuffle reduce, sigmoid
// ---------------------------------------------------------------------------
__global__ void edge_score_kernel(const float* __restrict__ fs, const float* __restrict__ fd,
                                  const int* __restrict__ src, const int* __restrict__ dst,
                                  float* __restrict__ out, int E) {
  int wid  = (int)((blockIdx.x * blockDim.x + threadIdx.x) >> 5);
  int lane = threadIdx.x & 31;
  if (wid >= E) return;
  const float* a = fs + (size_t)src[wid] * 256;
  const float* b = fd + (size_t)dst[wid] * 256;
  float acc = 0.0f;
#pragma unroll
  for (int i = 0; i < 8; ++i) acc += a[lane + 32 * i] * b[lane + 32 * i];
#pragma unroll
  for (int off = 16; off > 0; off >>= 1) acc += __shfl_xor(acc, off, 32);
  if (lane == 0) out[wid] = 1.0f / (1.0f + expf(-acc));
}

// ---------------------------------------------------------------------------
// Host orchestration.
// Input order = JAX pytree flatten of setup_inputs() (dict keys sorted):
//  0..3   ei_uu, ei_uv, ei_vu, ei_vv                       (int32 [2,E])
//  4..6   attn1.{W1,b1,w2}      7..9  attn2.{W1,b1,w2}
//  10..13 bn1.u.{b,g}, bn1.v.{b,g}   14..17 bn2 same
//  18..41 l1.{uu,uv,vu,vv}.{W_dst,W_src,W_upd,b_dst,b_src,b_upd}
//  42..65 l2 same
//  66..69 post.u.{W,b}, post.v.{W,b}
//  70     x_u [32768,128]    71   x_v [32768,128]
// ---------------------------------------------------------------------------
extern "C" void kernel_launch(void* const* d_in, const int* in_sizes, int n_in,
                              void* d_out, int out_size, void* d_ws, size_t ws_size,
                              hipStream_t stream) {
  (void)n_in; (void)out_size; (void)ws_size;
  const int NN  = 32768;   // N_U == N_V
  const int FIN = 128, H = 256, AA = 64;
  const int E   = in_sizes[0] / 2;

  // ---- workspace bump allocator -------------------------------------------
  char*  base = (char*)d_ws;
  size_t off  = 0;
  auto alloc = [&](size_t bytes) -> void* {
    void* p = base + off;
    off = (off + bytes + 255) & ~(size_t)255;
    return p;
  };
  auto toBf16 = [&](const float* src, int n) -> u16b* {
    u16b* dstp = (u16b*)alloc((size_t)n * 2);
    f2bf_kernel<<<(n + 255) / 256, 256, 0, stream>>>(src, dstp, n);
    return dstp;
  };

  // ---- weights (convert to bf16 once per launch) --------------------------
  struct CW { const u16b *Wd, *Ws, *Wu; const float *bd, *bs, *bu; };
  CW cw[2][4];   // relation order (sorted): 0=uu 1=uv 2=vu 3=vv
  for (int l = 0; l < 2; ++l) {
    int F = (l == 0) ? FIN : H;
    for (int r = 0; r < 4; ++r) {
      int bidx = 18 + l * 24 + r * 6;
      cw[l][r].Wd = toBf16((const float*)d_in[bidx + 0], H * F);
      cw[l][r].Ws = toBf16((const float*)d_in[bidx + 1], H * F);
      cw[l][r].Wu = toBf16((const float*)d_in[bidx + 2], H * 2 * H);
      cw[l][r].bd = (const float*)d_in[bidx + 3];
      cw[l][r].bs = (const float*)d_in[bidx + 4];
      cw[l][r].bu = (const float*)d_in[bidx + 5];
    }
  }
  const u16b*  attW1[2] = { toBf16((const float*)d_in[4], AA * H),
                            toBf16((const float*)d_in[7], AA * H) };
  const float* attb1[2] = { (const float*)d_in[5], (const float*)d_in[8] };
  const float* attw2[2] = { (const float*)d_in[6], (const float*)d_in[9] };
  const float* bnb[2][2] = { { (const float*)d_in[10], (const float*)d_in[12] },
                             { (const float*)d_in[14], (const float*)d_in[16] } };
  const float* bng[2][2] = { { (const float*)d_in[11], (const float*)d_in[13] },
                             { (const float*)d_in[15], (const float*)d_in[17] } };
  const u16b*  postW[2] = { toBf16((const float*)d_in[66], H * H),
                            toBf16((const float*)d_in[68], H * H) };
  const float* postb[2] = { (const float*)d_in[67], (const float*)d_in[69] };

  // ---- node buffers -------------------------------------------------------
  u16b*  xb[2];                                  // bf16 node features (u, v)
  xb[0] = (u16b*)alloc((size_t)NN * H * 2);
  xb[1] = (u16b*)alloc((size_t)NN * H * 2);
  float* aggsum  = (float*)alloc((size_t)NN * H * 4);   // reused as comb
  float* cnt     = (float*)alloc((size_t)NN * 4);
  u16b*  aggb    = (u16b*)alloc((size_t)NN * H * 2);
  u16b*  concatb = (u16b*)alloc((size_t)NN * 2 * H * 2);
  u16b*  eB[4];
  for (int r = 0; r < 4; ++r) eB[r] = (u16b*)alloc((size_t)NN * H * 2);
  float* tbuf  = (float*)alloc((size_t)NN * AA * 4);
  float* x2f[2];
  x2f[0] = (float*)alloc((size_t)NN * H * 4);
  x2f[1] = (float*)alloc((size_t)NN * H * 4);
  float* zsum   = (float*)alloc(256);
  float* alpha  = (float*)alloc(256);
  float* bnMean = (float*)alloc(H * 4);
  float* bnVar  = (float*)alloc(H * 4);
  // feats alias e-buffers (free after attention of layer 2)
  float* feats[2] = { (float*)eB[0], (float*)eB[2] };

  auto zero = [&](float* p, int n) {
    zero_kernel<<<(n + 255) / 256, 256, 0, stream>>>(p, n);
  };
  auto gemm = [&](const u16b* A, int lda, const u16b* W, int ldw, const float* bias,
                  float* outF, u16b* outB, int ldo, int M, int Ncols, int K) {
    dim3 g(M / 128, Ncols / 64);
    gemm_bf16_kernel<<<g, 256, 0, stream>>>(A, lda, W, ldw, bias, outF, outB, ldo, K);
  };

  // ---- one GNN layer ------------------------------------------------------
  auto runLayer = [&](int l, const float* xfu, const float* xfv, int F) {
    const float* xf[2] = { xfu, xfv };
    for (int r = 0; r < 4; ++r) {               // 0=uu 1=uv 2=vu 3=vv
      int st = (r <= 1) ? 0 : 1;                // src: uu,uv from u; vu,vv from v
      int dt = (r == 0 || r == 2) ? 0 : 1;      // dst: uu,vu into u; uv,vv into v
      const int* src = (const int*)d_in[r];
      const int* dst = src + E;
      zero(aggsum, NN * F);
      zero(cnt, NN);
      scatter_add_kernel<<<E, 128, 0, stream>>>(xf[st], src, dst, aggsum, cnt, F);
      finalize_mean_kernel<<<(NN * F + 255) / 256, 256, 0, stream>>>(aggsum, cnt, aggb,
                                                                      NN * F, F);
      // owns -> concat[:, :H]
      gemm(xb[dt], F, cw[l][r].Wd, F, cw[l][r].bd, nullptr, concatb, 2 * H, NN, H, F);
      // msgs -> concat[:, H:]
      gemm(aggb, F, cw[l][r].Ws, F, cw[l][r].bs, nullptr, concatb + H, 2 * H, NN, H, F);
      // upd -> e_r
      gemm(concatb, 2 * H, cw[l][r].Wu, 2 * H, cw[l][r].bu, nullptr, eB[r], H, NN, H,
           2 * H);
    }
    // attention + BN per node type  (u: [e_vu, e_uu] ; v: [e_uv, e_vv])
    for (int typ = 0; typ < 2; ++typ) {
      const u16b* em[2];
      if (typ == 0) { em[0] = eB[2]; em[1] = eB[0]; }
      else          { em[0] = eB[1]; em[1] = eB[3]; }
      zero(zsum, 2);
      for (int m = 0; m < 2; ++m) {
        gemm(em[m], H, attW1[l], H, attb1[l], tbuf, nullptr, AA, NN, AA, H);
        attn_z_kernel<<<NN / 256, 256, 0, stream>>>(tbuf, attw2[l], zsum + m, NN);
      }
      attn_alpha_kernel<<<1, 1, 0, stream>>>(zsum, 1.0f / NN, alpha);
      float* comb = aggsum;   // reuse
      attn_combine_kernel<<<(NN * H + 255) / 256, 256, 0, stream>>>(em[0], em[1], alpha,
                                                                     comb, NN * H);
      bn_stats_kernel<<<H, 256, 0, stream>>>(comb, bnMean, bnVar, NN, H);
      bn_apply_kernel<<<(NN * H + 255) / 256, 256, 0, stream>>>(
          comb, bnMean, bnVar, bng[l][typ], bnb[l][typ], x2f[typ], xb[typ], NN * H, H);
    }
  };

  // ---- pipeline -----------------------------------------------------------
  f2bf_kernel<<<(NN * FIN + 255) / 256, 256, 0, stream>>>((const float*)d_in[70], xb[0],
                                                          NN * FIN);
  f2bf_kernel<<<(NN * FIN + 255) / 256, 256, 0, stream>>>((const float*)d_in[71], xb[1],
                                                          NN * FIN);
  runLayer(0, (const float*)d_in[70], (const float*)d_in[71], FIN);
  runLayer(1, x2f[0], x2f[1], H);

  // post linear (inputs: BN2 bf16 outputs in xb[]) -> feats (aliases eB)
  for (int typ = 0; typ < 2; ++typ)
    gemm(xb[typ], H, postW[typ], H, postb[typ], feats[typ], nullptr, H, NN, H, H);

  // edge heads: output rows in reference order [uv, vu, uu, vv]
  float* out = (float*)d_out;
  struct RR { int ein, s, d; };
  RR rr[4] = { {1, 0, 1}, {2, 1, 0}, {0, 0, 0}, {3, 1, 1} };
  int blocks = (E * 32 + 255) / 256;
  for (int k = 0; k < 4; ++k) {
    const int* src = (const int*)d_in[rr[k].ein];
    const int* dst = src + E;
    edge_score_kernel<<<blocks, 256, 0, stream>>>(feats[rr[k].s], feats[rr[k].d], src,
                                                  dst, out + (size_t)k * E, E);
  }
}
